// DifferenceAttention_38835094291123
// MI455X (gfx1250) — matrix operand order repaired and checked by reading, not compiled, not run on hardware
//
#include <hip/hip_runtime.h>
#include <math.h>

typedef __attribute__((ext_vector_type(16))) _Float16 v16h;
typedef __attribute__((ext_vector_type(8)))  float    v8f;

#define LDIM  8192
#define NB    32
#define EDIM  64

// ---------------- TCN stage 1: t1 = relu(conv3(in)) -----------------
__global__ void k_conv3_relu(const float* __restrict__ in, long sB, long sC, long sL,
                             int Cin, const float* __restrict__ w,
                             const float* __restrict__ bias,
                             float* __restrict__ out, int Cout, int dil)
{
  int bc = blockIdx.y;
  int b = bc / Cout, co = bc % Cout;
  int l = blockIdx.x * blockDim.x + threadIdx.x;
  const float* wp = w + (long)co * Cin * 3;
  const float* ip = in + (long)b * sB;
  float acc = bias[co];
  for (int ci = 0; ci < Cin; ++ci) {
    long base = (long)ci * sC;
    int l0 = l - 2 * dil, l1 = l - dil;
    if (l0 >= 0) acc += wp[ci*3+0] * ip[base + (long)l0 * sL];
    if (l1 >= 0) acc += wp[ci*3+1] * ip[base + (long)l1 * sL];
    acc += wp[ci*3+2] * ip[base + (long)l * sL];
  }
  out[((long)b * Cout + co) * LDIM + l] = fmaxf(acc, 0.0f);
}

// ------- TCN stage 2: out = relu(relu(conv3(t1)) + conv1x1(h)) [+PE] -------
__global__ void k_conv3_res_relu(const float* __restrict__ t1,
                                 const float* __restrict__ w2, const float* __restrict__ b2,
                                 const float* __restrict__ hin, long sB, long sC, long sL, int Cin,
                                 const float* __restrict__ wd, const float* __restrict__ bd,
                                 float* __restrict__ out, int Cout, int dil, int add_pe)
{
  int bc = blockIdx.y;
  int b = bc / Cout, co = bc % Cout;
  int l = blockIdx.x * blockDim.x + threadIdx.x;
  const float* wp = w2 + (long)co * Cout * 3;
  const float* tp = t1 + (long)b * Cout * LDIM;
  float acc = b2[co];
  for (int ci = 0; ci < Cout; ++ci) {
    long base = (long)ci * LDIM;
    int l0 = l - 2 * dil, l1 = l - dil;
    if (l0 >= 0) acc += wp[ci*3+0] * tp[base + l0];
    if (l1 >= 0) acc += wp[ci*3+1] * tp[base + l1];
    acc += wp[ci*3+2] * tp[base + l];
  }
  acc = fmaxf(acc, 0.0f);           // relu on conv2 output
  float res = bd[co];
  const float* hp = hin + (long)b * sB;
  const float* wdp = wd + (long)co * Cin;
  for (int ci = 0; ci < Cin; ++ci)
    res += wdp[ci] * hp[(long)ci * sC + (long)l * sL];
  float v = fmaxf(acc + res, 0.0f); // relu(y + res)
  if (add_pe) {                     // positional encoding indexed by batch (source bug)
    int j = co >> 1;
    float div = __expf((float)(2 * j) * (-9.210340371976184f / 64.0f));
    float arg = (float)b * div;
    v += (co & 1) ? __cosf(arg) : __sinf(arg);
  }
  out[((long)b * Cout + co) * LDIM + l] = v;
}

// ---------------- zero small accumulators -----------------
__global__ void k_zero(float* p, long n)
{
  long i = (long)blockIdx.x * blockDim.x + threadIdx.x;
  if (i < n) p[i] = 0.0f;
}

// ---------------- Gram: G_b = ex_b * ex_b^T  (WMMA f16, atomic partials) ----
__global__ void k_gram(const float* __restrict__ ex, float* __restrict__ G, int lchunks)
{
  // grid (lchunks, 16 tile-pairs, NB), blockDim = 32 (one wave)
  int b = blockIdx.z;
  int tij = blockIdx.y, ti = tij >> 2, tj = tij & 3;
  int Lc = LDIM / lchunks;
  int lbase = blockIdx.x * Lc;
  int lane = threadIdx.x;
  int hi = (lane >= 16) ? 1 : 0;
  int m = lane & 15;
  const float* exb = ex + (long)b * EDIM * LDIM;
  const float* arow = exb + (long)(ti * 16 + m) * LDIM;
  const float* brow = exb + (long)(tj * 16 + m) * LDIM;
  v8f acc = {};
  for (int l0 = lbase; l0 < lbase + Lc; l0 += 32) {
    v16h a, bf;
#pragma unroll
    for (int h = 0; h < 16; ++h) {
      int ka = h + (hi ? 16 : 0);   // A 16x32 f16 lane layout
      a[h]  = (_Float16)arow[l0 + ka];
      int kb = h + (hi ? 16 : 0);                       // B 32x16 f16 lane layout
      bf[h] = (_Float16)brow[l0 + kb];
    }
    acc = __builtin_amdgcn_wmma_f32_16x16x32_f16(false, a, false, bf,
                                                 (short)0, acc, false, false);
  }
  float* g = G + ((long)b * EDIM + ti * 16) * EDIM + tj * 16;
#pragma unroll
  for (int r = 0; r < 8; ++r) {
    int mm = r + (hi ? 8 : 0);
    atomicAdd(&g[mm * EDIM + m], acc[r]);
  }
}

// ---------------- s_b = ex_b @ 1 (row sums over L) -----------------
__global__ void k_rowsum(const float* __restrict__ ex, float* __restrict__ s)
{
  __shared__ float red[256];
  int be = blockIdx.x;
  const float* p = ex + (long)be * LDIM;
  float acc = 0.0f;
  for (int l = threadIdx.x; l < LDIM; l += 256) acc += p[l];
  red[threadIdx.x] = acc;
  __syncthreads();
  for (int st = 128; st > 0; st >>= 1) {
    if (threadIdx.x < st) red[threadIdx.x] += red[threadIdx.x + st];
    __syncthreads();
  }
  if (threadIdx.x == 0) s[be] = red[0];
}

// ---- per-batch: scores softmax, M = A*Wv (f16), c = A*bv, x_avg path ----
__global__ void k_attn_small(const float* __restrict__ G, const float* __restrict__ s,
                             const float* wq, const float* bq,
                             const float* wk, const float* bk,
                             const float* wv, const float* bv,
                             const float* cw1, const float* cb1, const float* cg1, const float* cbe1,
                             const float* cw2, const float* cb2, const float* cg2, const float* cbe2,
                             _Float16* __restrict__ Mh, float* __restrict__ cvec,
                             float* __restrict__ xavg)
{
  __shared__ float tmp[EDIM * EDIM];
  __shared__ float ks_sh[EDIM], cm_sh[EDIM], a1_sh[16];
  int b = blockIdx.x, e = threadIdx.x;
  const float* Gb = G + (long)b * EDIM * EDIM;
  const float* sb = s + b * EDIM;
  // tmp[e][v] = sum_u Wq[e][u] * G[u][v]
  for (int v = 0; v < EDIM; ++v) {
    float acc = 0.0f;
    for (int u = 0; u < EDIM; ++u) acc += wq[e * EDIM + u] * Gb[u * EDIM + v];
    tmp[e * EDIM + v] = acc;
  }
  float qs = 0.0f, ks = 0.0f;
  for (int u = 0; u < EDIM; ++u) { qs += wq[e*EDIM+u] * sb[u]; ks += wk[e*EDIM+u] * sb[u]; }
  ks_sh[e] = ks;
  __syncthreads();
  // scores_pre row e, softmax over f
  float row[EDIM];
  float bqe = bq[e], mx = -1e30f;
  for (int f = 0; f < EDIM; ++f) {
    float acc = 0.0f;
    for (int v = 0; v < EDIM; ++v) acc += tmp[e * EDIM + v] * wk[f * EDIM + v];
    acc += qs * bk[f] + bqe * ks_sh[f] + (float)LDIM * bqe * bk[f];
    row[f] = acc;
    mx = fmaxf(mx, acc);
  }
  float sum = 0.0f;
  for (int f = 0; f < EDIM; ++f) { row[f] = __expf(row[f] - mx); sum += row[f]; }
  float inv = 1.0f / sum;
  for (int f = 0; f < EDIM; ++f) row[f] *= inv;
  // M row, c, cam_mean
  float ce = 0.0f, cm = 0.0f;
  for (int f = 0; f < EDIM; ++f) ce += row[f] * bv[f];
  for (int g = 0; g < EDIM; ++g) {
    float mg = 0.0f;
    for (int f = 0; f < EDIM; ++f) mg += row[f] * wv[f * EDIM + g];
    Mh[((long)b * EDIM + e) * EDIM + g] = (_Float16)mg;
    cm += mg * (sb[g] / (float)LDIM);
  }
  cvec[b * EDIM + e] = ce;
  cm_sh[e] = cm + ce;
  __syncthreads();
  float bnscale = rsqrtf(1.0f + 1e-5f);
  if (e < 16) {
    float acc = cb1[e];
    for (int u = 0; u < EDIM; ++u) acc += cw1[e * EDIM + u] * cm_sh[u];
    a1_sh[e] = fmaxf(acc * (cg1[e] * bnscale) + cbe1[e], 0.0f);
  }
  __syncthreads();
  float acc = cb2[e];
  for (int j = 0; j < 16; ++j) acc += cw2[e * 16 + j] * a1_sh[j];
  xavg[b * EDIM + e] = acc * (cg2[e] * bnscale) + cbe2[e];
}

// ---------------- cam_in = M_b @ ex + c_b  (WMMA f16) -----------------
__global__ void k_camgemm(const _Float16* __restrict__ Mh, const float* __restrict__ cvec,
                          const float* __restrict__ ex, float* __restrict__ cam)
{
  // grid (LDIM/16, NB), blockDim 32: one wave -> one 64x16 output tile
  int b = blockIdx.y;
  int l0 = blockIdx.x * 16;
  int lane = threadIdx.x;
  int hi = (lane >= 16) ? 1 : 0;
  int m = lane & 15;
  const _Float16* Mb = Mh + (long)b * EDIM * EDIM;
  const float* exb = ex + (long)b * EDIM * LDIM;
  v8f acc[4] = {};
#pragma unroll
  for (int kc = 0; kc < 2; ++kc) {
    v16h bf;
#pragma unroll
    for (int h = 0; h < 16; ++h) {
      int k = h + (hi ? 16 : 0);
      bf[h] = (_Float16)exb[(long)(kc * 32 + k) * LDIM + l0 + m];
    }
#pragma unroll
    for (int i = 0; i < 4; ++i) {
      v16h a;
      const _Float16* arow = Mb + (long)(i * 16 + m) * EDIM + kc * 32;
#pragma unroll
      for (int h = 0; h < 16; ++h) {
        int ka = h + (hi ? 16 : 0);
        a[h] = arow[ka];
      }
      acc[i] = __builtin_amdgcn_wmma_f32_16x16x32_f16(false, a, false, bf,
                                                      (short)0, acc[i], false, false);
    }
  }
#pragma unroll
  for (int i = 0; i < 4; ++i) {
#pragma unroll
    for (int r = 0; r < 8; ++r) {
      int mm = i * 16 + r + (hi ? 8 : 0);
      cam[((long)b * EDIM + mm) * LDIM + l0 + m] = acc[i][r] + cvec[b * EDIM + mm];
    }
  }
}

// -------- fused per-position epilogue: CAM gate, TAFF, l1/l2/l3, sigmoid ----
__global__ void k_epilogue(const float* __restrict__ cam, const float* __restrict__ ex,
                           const float* __restrict__ xavg,
                           const float* w3, const float* b3, const float* g3, const float* be3,
                           const float* w4, const float* b4, const float* g4, const float* be4,
                           const float* w1, const float* b1,
                           const float* w2, const float* b2,
                           const float* w3l, const float* b3l,
                           float* __restrict__ out)
{
  __shared__ float W3[16 * 64], B3[16], S3[16], Be3[16];
  __shared__ float W4[64 * 16], B4[64], S4[64], Be4[64];
  __shared__ float W1[32 * 64], B1[32];
  __shared__ float W2[16 * 32], B2[16];
  __shared__ float W3L[16];
  int t = threadIdx.x;
  float bnscale = rsqrtf(1.0f + 1e-5f);
  for (int i = t; i < 16 * 64; i += 256) { W3[i] = w3[i]; W4[i] = w4[i]; }
  for (int i = t; i < 32 * 64; i += 256) W1[i] = w1[i];
  for (int i = t; i < 16 * 32; i += 256) W2[i] = w2[i];
  if (t < 16) { B3[t] = b3[t]; S3[t] = g3[t] * bnscale; Be3[t] = be3[t];
                B2[t] = b2[t]; W3L[t] = w3l[t]; }
  if (t < 64) { B4[t] = b4[t]; S4[t] = g4[t] * bnscale; Be4[t] = be4[t]; }
  if (t < 32) B1[t] = b1[t];
  __syncthreads();
  long idx = (long)blockIdx.x * 256 + t;
  int b = (int)(idx / LDIM), l = (int)(idx % LDIM);
  const float* camc = cam + (long)b * EDIM * LDIM + l;
  const float* exc  = ex  + (long)b * EDIM * LDIM + l;
  const float* xav  = xavg + b * EDIM;
  float cv[64];
#pragma unroll
  for (int ch = 0; ch < 64; ++ch) cv[ch] = camc[(long)ch * LDIM];
  float xc1[16];
#pragma unroll
  for (int j = 0; j < 16; ++j) {
    float acc = B3[j];
    for (int ch = 0; ch < 64; ++ch) acc += W3[j * 64 + ch] * cv[ch];
    xc1[j] = fmaxf(acc * S3[j] + Be3[j], 0.0f);
  }
  float o1[32];
#pragma unroll
  for (int j = 0; j < 32; ++j) o1[j] = B1[j];
  for (int ch = 0; ch < 64; ++ch) {
    float acc = B4[ch];
#pragma unroll
    for (int j = 0; j < 16; ++j) acc += W4[ch * 16 + j] * xc1[j];
    float gate = 1.0f / (1.0f + __expf(-(acc * S4[ch] + Be4[ch] + xav[ch])));
    float a = gate * cv[ch] * exc[(long)ch * LDIM];         // cam_out * ex
    float taff = 0.5f * (a + (1.0f - a));                   // faithful merge (== ~0.5)
#pragma unroll
    for (int j = 0; j < 32; ++j) o1[j] += W1[j * 64 + ch] * taff;
  }
  float o2[16];
#pragma unroll
  for (int j = 0; j < 16; ++j) {
    float acc = B2[j];
    for (int k2 = 0; k2 < 32; ++k2) acc += W2[j * 32 + k2] * o1[k2];
    o2[j] = acc;
  }
  float o3 = b3l[0];
#pragma unroll
  for (int j = 0; j < 16; ++j) o3 += W3L[j] * o2[j];
  out[idx] = 1.0f / (1.0f + __expf(-o3));
}

extern "C" void kernel_launch(void* const* d_in, const int* in_sizes, int n_in,
                              void* d_out, int out_size, void* d_ws, size_t ws_size,
                              hipStream_t stream)
{
  const float* P[64];
  for (int i = 0; i < n_in && i < 64; ++i) P[i] = (const float*)d_in[i];
  const float* x = P[0];
  // tcn[i]: w1,b1,w2,b2,wd,bd at 1+6*i
  const float *t0w1=P[1],*t0b1=P[2],*t0w2=P[3],*t0b2=P[4],*t0wd=P[5],*t0bd=P[6];
  const float *t1w1=P[7],*t1b1=P[8],*t1w2=P[9],*t1b2=P[10],*t1wd=P[11],*t1bd=P[12];
  const float *t2w1=P[13],*t2b1=P[14],*t2w2=P[15],*t2b2=P[16],*t2wd=P[17],*t2bd=P[18];
  const float *qw=P[19],*qb=P[20],*kw=P[21],*kb=P[22],*v1w=P[23],*v1b=P[24];
  // P[25],P[26] = v2 (unused: reference feeds v1 weights to both paths)
  const float *cw1=P[27],*cb1=P[28],*cg1=P[29],*cbe1=P[30];
  const float *cw2=P[31],*cb2=P[32],*cg2=P[33],*cbe2=P[34];
  const float *cw3=P[35],*cb3=P[36],*cg3=P[37],*cbe3=P[38];
  const float *cw4=P[39],*cb4=P[40],*cg4=P[41],*cbe4=P[42];
  const float *l1w=P[43],*l1b=P[44],*l2w=P[45],*l2b=P[46],*l3w=P[47],*l3b=P[48];

  char* ws = (char*)d_ws;
  const size_t REG = (size_t)NB * EDIM * LDIM * sizeof(float); // 64 MB
  float* A  = (float*)ws;              // b0_t1 -> h1 -> cam_in
  float* Bb = (float*)(ws + REG);      // h0 -> ex
  float* C  = (float*)(ws + 2 * REG);  // b1_t1 -> b2_t1
  float* G    = (float*)(ws + 3 * REG);
  float* svec = G + NB * EDIM * EDIM;
  _Float16* Mh = (_Float16*)(svec + NB * EDIM);
  float* cvec = (float*)(Mh + NB * EDIM * EDIM);
  float* xavg = cvec + NB * EDIM;

  dim3 blk(256);
  // ---- TCN block 0 (10 -> 16, dil 1); x is [B, L, 10] channel-last ----
  k_conv3_relu<<<dim3(LDIM/256, NB*16), blk, 0, stream>>>(
      x, (long)LDIM*10, 1L, 10L, 10, t0w1, t0b1, A, 16, 1);
  k_conv3_res_relu<<<dim3(LDIM/256, NB*16), blk, 0, stream>>>(
      A, t0w2, t0b2, x, (long)LDIM*10, 1L, 10L, 10, t0wd, t0bd, Bb, 16, 1, 0);
  // ---- TCN block 1 (16 -> 32, dil 2) ----
  k_conv3_relu<<<dim3(LDIM/256, NB*32), blk, 0, stream>>>(
      Bb, (long)16*LDIM, (long)LDIM, 1L, 16, t1w1, t1b1, C, 32, 2);
  k_conv3_res_relu<<<dim3(LDIM/256, NB*32), blk, 0, stream>>>(
      C, t1w2, t1b2, Bb, (long)16*LDIM, (long)LDIM, 1L, 16, t1wd, t1bd, A, 32, 2, 0);
  // ---- TCN block 2 (32 -> 64, dil 4), fuse positional encoding -> ex ----
  k_conv3_relu<<<dim3(LDIM/256, NB*64), blk, 0, stream>>>(
      A, (long)32*LDIM, (long)LDIM, 1L, 32, t2w1, t2b1, C, 64, 4);
  k_conv3_res_relu<<<dim3(LDIM/256, NB*64), blk, 0, stream>>>(
      C, t2w2, t2b2, A, (long)32*LDIM, (long)LDIM, 1L, 32, t2wd, t2bd, Bb, 64, 4, 1);
  // ---- attention via Gram refactor ----
  long nz = (long)NB*EDIM*EDIM + NB*EDIM;
  k_zero<<<dim3((unsigned)((nz + 255) / 256)), blk, 0, stream>>>(G, nz);
  k_gram<<<dim3(8, 16, NB), dim3(32), 0, stream>>>(Bb, G, 8);
  k_rowsum<<<dim3(NB*EDIM), blk, 0, stream>>>(Bb, svec);
  k_attn_small<<<dim3(NB), dim3(64), 0, stream>>>(
      G, svec, qw, qb, kw, kb, v1w, v1b,
      cw1, cb1, cg1, cbe1, cw2, cb2, cg2, cbe2, Mh, cvec, xavg);
  k_camgemm<<<dim3(LDIM/16, NB), dim3(32), 0, stream>>>(Mh, cvec, Bb, A);
  // ---- fused CAM-gate / TAFF / head ----
  k_epilogue<<<dim3(NB*LDIM/256), blk, 0, stream>>>(
      A, Bb, xavg, cw3, cb3, cg3, cbe3, cw4, cb4, cg4, cbe4,
      l1w, l1b, l2w, l2b, l3w, l3b, (float*)d_out);
}
